// MultiWayMLP_80925773791341
// MI455X (gfx1250) — matrix operand order repaired and checked by reading, not compiled, not run on hardware
//
#include <hip/hip_runtime.h>
#include <stdint.h>

// ---------------- problem constants ----------------
#define DD 1024
#define HH 4096
#define NTOK 16384          // B*N = 2*8192
#define MT 64               // token rows per workgroup tile
#define NSLOT 16512         // NTOK + 2*MT worst-case partition padding
#define NTILE (NSLOT / MT)  // 258
#define HC 128              // H chunk width

typedef __attribute__((ext_vector_type(16))) __bf16 v16bf;
typedef __attribute__((ext_vector_type(8)))  float  v8f;
typedef __attribute__((ext_vector_type(4)))  unsigned int u32x4;
typedef __attribute__((ext_vector_type(8)))  int  i32x8;
typedef __attribute__((ext_vector_type(4)))  int  i32x4;

#if defined(__has_builtin)
# if __has_builtin(__builtin_amdgcn_tensor_load_to_lds)
#  define USE_TDM 1
# endif
#endif

union ABFrag { int4 q[2]; v16bf v; };

__device__ __forceinline__ unsigned short f2bf(float f) {
    unsigned u = __float_as_uint(f);
    u += 0x7fffu + ((u >> 16) & 1u);   // round-to-nearest-even
    return (unsigned short)(u >> 16);
}
__device__ __forceinline__ float gelu_exact(float v) {
    return 0.5f * v * (1.0f + erff(v * 0.70710678118654752f));
}

// ---------------- workspace layout (bytes) ----------------
#define WS_HDR_OFF   0
#define WS_TOK_OFF   64                                   // int[NSLOT]
#define WS_XLN_OFF   66560                                 // bf16[NSLOT*DD]
#define WS_W1T_OFF   (WS_XLN_OFF + (size_t)NSLOT*DD*2)     // bf16[2][HH][DD]
#define WS_W2T_OFF   (WS_W1T_OFF + (size_t)2*HH*DD*2)      // bf16[2][DD][HH]
#define WS_TOTAL     (WS_W2T_OFF + (size_t)2*DD*HH*2)

// ================= 1) deterministic partition (single WG scan) =============
__global__ __launch_bounds__(1024)
void partition_kernel(const unsigned char* __restrict__ mask,
                      int* __restrict__ tok_of_slot, int* __restrict__ hdr) {
    __shared__ int sc[1024];
    __shared__ int s_total;
    const int tid = threadIdx.x;
    for (int s = tid; s < NSLOT; s += 1024) tok_of_slot[s] = -1;

    const int base = tid * 16;
    unsigned char mloc[16];
    int ones = 0;
    for (int i = 0; i < 16; ++i) { mloc[i] = mask[base + i]; ones += (mloc[i] != 0); }
    sc[tid] = ones;
    __syncthreads();
    for (int off = 1; off < 1024; off <<= 1) {        // inclusive Hillis-Steele
        int v = (tid >= off) ? sc[tid - off] : 0;
        __syncthreads();
        sc[tid] += v;
        __syncthreads();
    }
    const int incl = sc[tid];
    const int excl = incl - ones;
    if (tid == 1023) s_total = incl;
    __syncthreads();
    const int total = s_total;
    const int S = ((total + MT - 1) / MT) * MT;       // 64-aligned expr region start
    if (tid == 0) hdr[0] = S;

    int oneOff = excl;
    int zeroOff = base - excl;
    for (int i = 0; i < 16; ++i) {
        int t = base + i;
        int slot = mloc[i] ? (oneOff++) : (S + zeroOff++);
        tok_of_slot[slot] = t;
    }
}

// ================= 2) f32 -> bf16 transpose of weights =====================
// dst[c*rows + r] = bf16(src[r*cols + c])
__global__ __launch_bounds__(256)
void cvt_transpose_kernel(const float* __restrict__ src,
                          unsigned short* __restrict__ dst, int rows, int cols) {
    int id = blockIdx.x * 256 + threadIdx.x;
    if (id >= rows * cols) return;
    int r = id / cols, c = id % cols;
    dst[(size_t)c * rows + r] = f2bf(src[id]);
}

// ================= 3) LayerNorm into compacted slot order ==================
__global__ __launch_bounds__(256)
void ln_kernel(const float* __restrict__ x, const unsigned char* __restrict__ mask,
               const float* __restrict__ g_rgb, const float* __restrict__ b_rgb,
               const float* __restrict__ g_expr, const float* __restrict__ b_expr,
               const int* __restrict__ tok_of_slot, unsigned short* __restrict__ xln) {
    const int slot = blockIdx.x;
    const int tid  = threadIdx.x;
    unsigned short* dst = xln + (size_t)slot * DD;
    const int t = tok_of_slot[slot];
    if (t < 0) {                               // zero-pad unused rows
        uint2 z; z.x = 0u; z.y = 0u;
        *(uint2*)(dst + tid * 4) = z;
        return;
    }
    const float* xr = x + (size_t)t * DD;
    float v[4], s = 0.f, sq = 0.f;
    for (int k = 0; k < 4; ++k) {
        v[k] = xr[tid + 256 * k];
        s += v[k]; sq += v[k] * v[k];
    }
    for (int m = 16; m >= 1; m >>= 1) {        // wave32 reduce
        s  += __shfl_xor(s,  m, 32);
        sq += __shfl_xor(sq, m, 32);
    }
    __shared__ float ws_[8], wq_[8];
    const int wid = tid >> 5, lane = tid & 31;
    if (lane == 0) { ws_[wid] = s; wq_[wid] = sq; }
    __syncthreads();
    if (tid == 0) {
        float a = 0.f, b = 0.f;
        for (int i = 0; i < 8; ++i) { a += ws_[i]; b += wq_[i]; }
        ws_[0] = a; wq_[0] = b;
    }
    __syncthreads();
    const float mu  = ws_[0] * (1.0f / DD);
    const float var = wq_[0] * (1.0f / DD) - mu * mu;
    const float rs  = rsqrtf(var + 1e-5f);
    const bool rgb = (mask[t] != 0);
    const float* g = rgb ? g_rgb : g_expr;
    const float* b = rgb ? b_rgb : b_expr;
    for (int k = 0; k < 4; ++k) {
        int d = tid + 256 * k;
        dst[d] = f2bf((v[k] - mu) * rs * g[d] + b[d]);
    }
}

// ================= 4) fused MLP: GEMM1 -> GELU -> GEMM2 (WMMA bf16) ========
__global__ __launch_bounds__(512)
void mlp_kernel(const unsigned short* __restrict__ xln,
                const unsigned short* __restrict__ w1t,   // [2][HH][DD] bf16
                const unsigned short* __restrict__ w2t,   // [2][DD][HH] bf16
                const float* __restrict__ b1_rgb, const float* __restrict__ b1_expr,
                const float* __restrict__ b2_rgb, const float* __restrict__ b2_expr,
                const int* __restrict__ tok_of_slot, const int* __restrict__ hdr,
                _Float16* __restrict__ out) {
    extern __shared__ char smem[];
    unsigned short* lds_x = (unsigned short*)smem;     // [MT][DD]  128 KB
    unsigned short* lds_h = lds_x + MT * DD;           // [MT][HC]   16 KB

    const int tid  = threadIdx.x;
    const int wave = tid >> 5, lane = tid & 31;
    const int mi = wave & 3, ng = wave >> 2;           // 4 M-tiles x 4 N-groups
    const int lan = lane & 15, sel = lane >> 4;
    const int tilebase = blockIdx.x * MT;

    const int S = hdr[0];
    const int e = (tilebase >= S) ? 1 : 0;
    const unsigned short* w1 = w1t + (size_t)e * HH * DD;
    const unsigned short* w2 = w2t + (size_t)e * DD * HH;
    const float* b1 = e ? b1_expr : b1_rgb;
    const float* b2 = e ? b2_expr : b2_rgb;

    // ---- stage the 64x1024 bf16 activation tile into LDS ----
#ifdef USE_TDM
    // Tensor Data Mover: one wave issues a single 2-D tile DMA (ISA ch.8 D#),
    // tracked by TENSORcnt; fence with s_wait_tensorcnt then barrier.
    if (wave == 0) {
        const unsigned lds_off = (unsigned)(uintptr_t)lds_x;   // addr[31:0] = LDS offset
        const unsigned long long ga =
            (unsigned long long)(uintptr_t)(xln + (size_t)tilebase * DD);
        u32x4 g0;
        g0[0] = 1u;                                  // count=1, is_restore=0, gather off
        g0[1] = lds_off;                             // lds_addr
        g0[2] = (unsigned)ga;                        // global_addr[31:0]
        g0[3] = ((unsigned)(ga >> 32) & 0x1FFFFFFu)  // global_addr[56:32]
              | (2u << 30);                          // type=2 ("image")
        i32x8 g1;
        g1[0] = (int)(1u << 16);                     // wg_mask=0, data_size=1 (2 bytes)
        g1[1] = (int)((unsigned)DD << 16);           // tensor_dim0[15:0]=1024 (bits 63:48)
        g1[2] = (int)((unsigned)(NSLOT & 0xffff) << 16); // tensor_dim1[15:0] (bits 95:80)
        g1[3] = (int)((unsigned)DD << 16);           // tile_dim0=1024 (bits 127:112)
        g1[4] = MT;                                  // tile_dim1=64, tile_dim2=0
        g1[5] = DD;                                  // tensor_dim0_stride[31:0]=1024
        g1[6] = 0;                                   // stride high bits / dim1_stride lo
        g1[7] = 0;
        i32x4 g2 = {0, 0, 0, 0};
        i32x4 g3 = {0, 0, 0, 0};
#if __clang_major__ >= 23
        i32x8 gx = {0, 0, 0, 0, 0, 0, 0, 0};
        __builtin_amdgcn_tensor_load_to_lds(g0, g1, g2, g3, gx, 0);
#else
        __builtin_amdgcn_tensor_load_to_lds(g0, g1, g2, g3, 0);
#endif
        __builtin_amdgcn_s_wait_tensorcnt(0);
    }
#else
    {
        const int4* src = (const int4*)(xln + (size_t)tilebase * DD);
        int4* dst = (int4*)lds_x;
        for (int i = tid; i < MT * DD / 8; i += 512) dst[i] = src[i];
    }
#endif
    __syncthreads();

    const v8f vzero = {0.f,0.f,0.f,0.f,0.f,0.f,0.f,0.f};
    v8f acc[16];
    #pragma unroll
    for (int i = 0; i < 16; ++i) acc[i] = vzero;

    const int m0 = mi * 16 + lan;                      // A-fragment row

    for (int hc = 0; hc < HH; hc += HC) {
        // ---- GEMM1: h[64x128] = Atile[64x1024] @ W1t^T chunk, K=1024
        v8f c1a = vzero, c1b = vzero;
        const int h0 = hc + (ng * 2 + 0) * 16 + lan;
        const int h1 = hc + (ng * 2 + 1) * 16 + lan;
        // pull next chunk's W1 rows toward the WGP (L2 resident)
        if (hc + HC < HH) {
            __builtin_prefetch(w1 + (size_t)(h0 + HC) * DD, 0, 3);
            __builtin_prefetch(w1 + (size_t)(h1 + HC) * DD, 0, 3);
        }
        for (int kb = 0; kb < DD; kb += 32) {
            ABFrag a;
            const unsigned short* ap = lds_x + m0 * DD + kb + sel * 16;
            a.q[0] = *(const int4*)ap;
            a.q[1] = *(const int4*)(ap + 8);
            ABFrag b0, bX;
            const unsigned short* bp0 = w1 + (size_t)h0 * DD + kb + sel * 16;
            const unsigned short* bp1 = w1 + (size_t)h1 * DD + kb + sel * 16;
            b0.q[0] = *(const int4*)bp0;  b0.q[1] = *(const int4*)(bp0 + 8);
            bX.q[0] = *(const int4*)bp1;  bX.q[1] = *(const int4*)(bp1 + 8);
            c1a = __builtin_amdgcn_wmma_f32_16x16x32_bf16(false, a.v, false, b0.v, (short)0, c1a, false, false);
            c1b = __builtin_amdgcn_wmma_f32_16x16x32_bf16(false, a.v, false, bX.v, (short)0, c1b, false, false);
        }
        // bias + exact GELU -> LDS h-block (bf16, row-major [MT][HC])
        const float bb0 = b1[h0], bb1 = b1[h1];
        #pragma unroll
        for (int i = 0; i < 8; ++i) {
            int row = mi * 16 + i + 8 * sel;
            lds_h[row * HC + (ng * 2 + 0) * 16 + lan] = f2bf(gelu_exact(c1a[i] + bb0));
            lds_h[row * HC + (ng * 2 + 1) * 16 + lan] = f2bf(gelu_exact(c1b[i] + bb1));
        }
        __syncthreads();

        // ---- GEMM2: z[64x1024] += h[64x128] @ W2 chunk, this wave: 16 N-tiles
        for (int k2 = 0; k2 < HC; k2 += 32) {
            ABFrag a2;
            const unsigned short* ap2 = lds_h + m0 * HC + k2 + sel * 16;
            a2.q[0] = *(const int4*)ap2;
            a2.q[1] = *(const int4*)(ap2 + 8);
            const int kk = hc + k2 + sel * 16;
            #pragma unroll
            for (int nt = 0; nt < 16; ++nt) {
                const int dcol = ng * 256 + nt * 16 + lan;
                ABFrag b;
                const unsigned short* bp = w2 + (size_t)dcol * HH + kk;
                b.q[0] = *(const int4*)bp;
                b.q[1] = *(const int4*)(bp + 8);
                acc[nt] = __builtin_amdgcn_wmma_f32_16x16x32_bf16(false, a2.v, false, b.v, (short)0, acc[nt], false, false);
            }
        }
        __syncthreads();
    }

    // epilogue: +b2, fp16, scatter back to original token rows
    int trow[8];
    #pragma unroll
    for (int i = 0; i < 8; ++i)
        trow[i] = tok_of_slot[tilebase + mi * 16 + i + 8 * sel];
    #pragma unroll
    for (int nt = 0; nt < 16; ++nt) {
        const int dcol = ng * 256 + nt * 16 + lan;
        const float bias = b2[dcol];
        #pragma unroll
        for (int i = 0; i < 8; ++i) {
            int t = trow[i];
            if (t >= 0) out[(size_t)t * DD + dcol] = (_Float16)(acc[nt][i] + bias);
        }
    }
}

// ================= launch ==================================================
extern "C" void kernel_launch(void* const* d_in, const int* in_sizes, int n_in,
                              void* d_out, int out_size, void* d_ws, size_t ws_size,
                              hipStream_t stream) {
    (void)in_sizes; (void)n_in; (void)out_size;
    const float*         x       = (const float*)d_in[0];
    const unsigned char* mask    = (const unsigned char*)d_in[1];   // jnp bool_ = 1 byte
    const float* g_rgb  = (const float*)d_in[2];
    const float* b_rgb  = (const float*)d_in[3];
    const float* w1_rgb = (const float*)d_in[4];
    const float* b1_rgb = (const float*)d_in[5];
    const float* w2_rgb = (const float*)d_in[6];
    const float* b2_rgb = (const float*)d_in[7];
    const float* g_expr  = (const float*)d_in[8];
    const float* b_expr  = (const float*)d_in[9];
    const float* w1_expr = (const float*)d_in[10];
    const float* b1_expr = (const float*)d_in[11];
    const float* w2_expr = (const float*)d_in[12];
    const float* b2_expr = (const float*)d_in[13];

    if (ws_size < WS_TOTAL) return;   // need ~67.5 MB of scratch
    char* ws = (char*)d_ws;
    int*            hdr = (int*)(ws + WS_HDR_OFF);
    int*            tok = (int*)(ws + WS_TOK_OFF);
    unsigned short* xln = (unsigned short*)(ws + WS_XLN_OFF);
    unsigned short* w1t = (unsigned short*)(ws + WS_W1T_OFF);
    unsigned short* w2t = (unsigned short*)(ws + WS_W2T_OFF);

    partition_kernel<<<1, 1024, 0, stream>>>(mask, tok, hdr);

    const int wgrid = (DD * HH + 255) / 256;
    cvt_transpose_kernel<<<wgrid, 256, 0, stream>>>(w1_rgb,  w1t,                    DD, HH);
    cvt_transpose_kernel<<<wgrid, 256, 0, stream>>>(w1_expr, w1t + (size_t)HH * DD,  DD, HH);
    cvt_transpose_kernel<<<wgrid, 256, 0, stream>>>(w2_rgb,  w2t,                    HH, DD);
    cvt_transpose_kernel<<<wgrid, 256, 0, stream>>>(w2_expr, w2t + (size_t)DD * HH,  HH, DD);

    ln_kernel<<<NSLOT, 256, 0, stream>>>(x, mask, g_rgb, b_rgb, g_expr, b_expr, tok, xln);

    const size_t smem_bytes = (size_t)(MT * DD + MT * HC) * 2;   // 144 KB
    mlp_kernel<<<NTILE, 512, smem_bytes, stream>>>(xln, w1t, w2t,
                                                   b1_rgb, b1_expr, b2_rgb, b2_expr,
                                                   tok, hdr, (_Float16*)d_out);
}
